// CNN_k5_two_layer_relu_mp_thresh_quantized_82016695485093
// MI455X (gfx1250) — compile-verified, matrix operand order checked by reading, not run on hardware
//
#include <hip/hip_runtime.h>
#include <stdint.h>

// ---------------------------------------------------------------------------
// Binary 2-layer CNN for MI455X (gfx1250, wave32).
//
// Per layer (fused in ONE kernel):
//   sign(in) -> 5x5 conv (1ch -> 16ch, sign weights) -> relu -> maxpool4x4
//            -> sign(out - thresh) -> 4x4 channel-block reshuffle
//
// Conv is mapped to V_WMMA_I32_16X16X64_IU8:
//   A (16x64 i8)  = sign weights, M = out-channel, K = 8*dy + dx (dx<5), rest 0
//   B (64x16 i8)  = image patches, N = output x pixel, same K packing
//   D (16x16 i32) = exact integer conv result for one y row, 16 x pixels
//
// K packing 8*dy+dx makes every B VGPR = 4 consecutive bytes of one LDS image
// row -> two aligned ds_load_b32 + funnel shift per row per lane, branchless.
//
// Pooling: acc = max over 4 y rows of max(D,0)  (relu folded into max-with-0),
// then x-pool with __shfl_xor over lane groups of 4 (lane = x pixel in D).
//
// Reshuffle: channel c = 4g+i -> output pixel (i*128+hh, g*128+ww).
// Layer1 writes +-1 int8 intermediate into d_ws (4MB, L2-resident);
// Layer2 writes +-1 float into d_out.
// ---------------------------------------------------------------------------

typedef int v8i __attribute__((ext_vector_type(8)));

#define IMG   512
#define TX    64          // tile width  (input pixels)
#define TY    32          // tile height (input pixels)
#define LDSW  72          // TX + 4 halo, padded to multiple of 4 bytes
#define LDSH  36          // TY + 4 halo

static __device__ __forceinline__ int sgnf(float x) {
    return (x > 0.0f) - (x < 0.0f);          // -1 / 0 / +1, matches jnp.sign
}

template <int LAYER>
__global__ __launch_bounds__(256)
void bconv_layer(const float*  __restrict__ xin,   // layer1: fp32 input image
                 const int8_t* __restrict__ bin,   // layer2: +-1 int8 image
                 const float*  __restrict__ w,     // [16][5][5] fp32 weights
                 const float*  __restrict__ thresh,// [1]
                 int8_t*       __restrict__ bout,  // layer1 output (int8)
                 float*        __restrict__ fout)  // layer2 output (fp32)
{
    __shared__ __align__(16) unsigned char tile[LDSH * LDSW];
    const uint32_t* tile32 = (const uint32_t*)tile;

    const int b    = blockIdx.z;
    const int gy0  = blockIdx.y * TY;
    const int gx0  = blockIdx.x * TX;
    const int tid  = threadIdx.x;
    const int lane = tid & 31;
    const int wv   = tid >> 5;                 // wave id, uniform per wave
    const int half = lane >> 4;                // lane 0-15 vs 16-31
    const float th = thresh[0];
    const size_t imgBase = (size_t)b * IMG * IMG;

    // ------------------------------------------------------------------
    // Build the A matrix (sign weights) in registers, once per lane.
    // A 16x64 i8 layout (wave32): lanes 0-15 hold M=0-15 with
    //   V0:K0-3 V1:K4-7 V2:K16-19 V3:K20-23 V4:K32-35 V5:K36-39 V6/7:K48+
    // lanes 16-31 hold M=0-15 with K 8-15, 24-31, 40-47, 56-63.
    // With K = 8*dy+dx, VGPR pair p covers dy = 2*p + half.
    // ------------------------------------------------------------------
    const int ch = lane & 15;
    v8i av;
    #pragma unroll
    for (int p = 0; p < 4; ++p) {
        const int dy  = 2 * p + half;
        const int dyc = (dy < 5) ? dy : 0;
        const uint32_t m = (dy < 5) ? 0xFFFFFFFFu : 0u;
        const float* wr = w + (ch * 25 + dyc * 5);
        uint32_t lo = ((uint32_t)(uint8_t)(int8_t)sgnf(wr[0]))       |
                      ((uint32_t)(uint8_t)(int8_t)sgnf(wr[1]) << 8)  |
                      ((uint32_t)(uint8_t)(int8_t)sgnf(wr[2]) << 16) |
                      ((uint32_t)(uint8_t)(int8_t)sgnf(wr[3]) << 24);
        uint32_t hi =  (uint32_t)(uint8_t)(int8_t)sgnf(wr[4]);       // dx=4
        av[2*p + 0] = (int)(lo & m);
        av[2*p + 1] = (int)(hi & m);
    }

    // ------------------------------------------------------------------
    // Cooperative LDS fill: +-1 int8 image tile with 2-pixel halo,
    // zero outside the image (== conv zero padding of the sign image).
    // ------------------------------------------------------------------
    for (int t = tid; t < LDSH * (TX + 4); t += 256) {
        const int row = t / (TX + 4);
        const int col = t - row * (TX + 4);
        const int gy  = gy0 - 2 + row;
        const int gx  = gx0 - 2 + col;
        int8_t v = 0;
        if (gy >= 0 && gy < IMG && gx >= 0 && gx < IMG) {
            const size_t o = imgBase + (size_t)gy * IMG + gx;
            v = (LAYER == 1) ? (int8_t)sgnf(xin[o]) : bin[o];
        }
        tile[row * LDSW + col] = (unsigned char)v;
    }
    __syncthreads();

    // ------------------------------------------------------------------
    // Each wave: 16x16 x/y sub-tile -> 4 pooled rows x 4 pooled cols x 16ch.
    // ------------------------------------------------------------------
    const int sx   = (wv & 3) * 16;            // x offset in tile
    const int syw  = (wv >> 2) * 16;           // y offset in tile
    const int coln = sx + (lane & 15);         // LDS col of dx=0 for this lane

    #pragma unroll
    for (int yb = 0; yb < 4; ++yb) {           // one pooled row per iteration
        v8i acc = {0, 0, 0, 0, 0, 0, 0, 0};    // relu folded in (max with 0)

        #pragma unroll
        for (int yd = 0; yd < 4; ++yd) {       // 4 y rows feeding this pool
            const int rbase = syw + yb * 4 + yd;   // + dy = LDS row index
            // B 64x16 i8 layout: lane = column N; lanes 0-15 K 0-7/16-23/32-39/48-55,
            // lanes 16-31 K 8-15/24-31/40-47/56-63; VGPR pair p covers dy = 2*p+half.
            v8i bv;
            #pragma unroll
            for (int p = 0; p < 4; ++p) {
                const int dy  = 2 * p + half;
                const int dyc = (dy < 5) ? dy : 0;
                const uint32_t m = (dy < 5) ? 0xFFFFFFFFu : 0u;
                const int boff = (rbase + dyc) * LDSW + coln;  // bytes 0..4
                const uint32_t w0 = tile32[boff >> 2];
                const uint32_t w1 = tile32[(boff >> 2) + 1];
                uint64_t v = (((uint64_t)w1 << 32) | w0) >> ((boff & 3) * 8);
                bv[2*p + 0] = (int)((uint32_t)v & m);           // dx 0..3
                bv[2*p + 1] = (int)(((uint32_t)(v >> 32) & 0xFFu) & m); // dx 4
            }
            v8i c0 = {0, 0, 0, 0, 0, 0, 0, 0};
            // signed i8 x signed i8 -> exact i32 binary conv dot products
            v8i d = __builtin_amdgcn_wmma_i32_16x16x64_iu8(
                        true, av, true, bv, c0, false, false);
            #pragma unroll
            for (int r = 0; r < 8; ++r)
                acc[r] = (d[r] > acc[r]) ? d[r] : acc[r];       // relu + y-max
        }

        // x-pool: lanes {4k..4k+3} are 4 consecutive x pixels
        #pragma unroll
        for (int r = 0; r < 8; ++r) {
            int v = acc[r];
            int v1 = __shfl_xor(v, 1, 32); v = (v1 > v) ? v1 : v;
            int v2 = __shfl_xor(v, 2, 32); v = (v2 > v) ? v2 : v;
            acc[r] = v;
        }

        // threshold-sign + reshuffle + store (one lane per pool group)
        if ((lane & 3) == 0) {
            const int hh = ((gy0 + syw) >> 2) + yb;                 // 0..127
            const int ww = ((gx0 + sx) >> 2) + ((lane & 15) >> 2);  // 0..127
            #pragma unroll
            for (int r = 0; r < 8; ++r) {
                const int c = r + 8 * half;        // D row M = out channel
                const int i = c & 3;               // c = 4g + i
                const int g = c >> 2;
                const float f = (float)acc[r] - th;
                const float s = (f > 0.0f) ? 1.0f : ((f < 0.0f) ? -1.0f : 0.0f);
                const size_t o = imgBase + (size_t)(i * 128 + hh) * IMG
                               + (size_t)(g * 128 + ww);
                if (LAYER == 1) bout[o] = (int8_t)s;
                else            fout[o] = s;
            }
        }
    }
}

extern "C" void kernel_launch(void* const* d_in, const int* in_sizes, int n_in,
                              void* d_out, int out_size, void* d_ws, size_t ws_size,
                              hipStream_t stream) {
    const float* x  = (const float*)d_in[0];   // [B,1,512,512] fp32
    const float* w1 = (const float*)d_in[1];   // [16,1,5,5]
    const float* w2 = (const float*)d_in[2];   // [16,1,5,5]
    const float* t1 = (const float*)d_in[3];   // [1]
    const float* t2 = (const float*)d_in[4];   // [1]
    float*  out = (float*)d_out;               // [B,1,512,512] fp32 (+-1)
    int8_t* mid = (int8_t*)d_ws;               // [B,512,512] int8 (+-1), ~4MB

    const int B = in_sizes[0] / (IMG * IMG);
    dim3 grid(IMG / TX, IMG / TY, B);          // (8, 16, B)
    dim3 block(256);

    bconv_layer<1><<<grid, block, 0, stream>>>(x, nullptr, w1, t1, mid, nullptr);
    bconv_layer<2><<<grid, block, 0, stream>>>(nullptr, mid, w2, t2, nullptr, out);
}
